// W2TransformerBlock_79078937854233
// MI455X (gfx1250) — matrix operand order repaired and checked by reading, not compiled, not run on hardware
//
#include <hip/hip_runtime.h>

#define AH   16
#define AD   128
#define ASEQ 2048
#define ACH  2048
#define AFF  8192
#define INV_2SQRTD 0.04419417382415922f   // 1 / (2*sqrt(128))

typedef __bf16 bf16;
typedef __attribute__((ext_vector_type(16))) __bf16 v16bf;
typedef __attribute__((ext_vector_type(8)))  __bf16 bf16x8;
typedef __attribute__((ext_vector_type(4)))  __bf16 bf16x4;
typedef __attribute__((ext_vector_type(8)))  float  v8f;
typedef __attribute__((ext_vector_type(4)))  float  f32x4;
typedef __attribute__((ext_vector_type(4)))  int    v4i;

#ifndef __has_builtin
#define __has_builtin(x) 0
#endif
#if __has_builtin(__builtin_amdgcn_global_load_async_to_lds_b128)
#define USE_ASYNC_LDS 1
typedef __attribute__((address_space(1))) v4i* as1_v4i;
typedef __attribute__((address_space(3))) v4i* as3_v4i;
#endif

static __device__ __forceinline__ v16bf mk_frag(bf16x8 lo, bf16x8 hi) {
  return __builtin_shufflevector(lo, hi, 0, 1, 2, 3, 4, 5, 6, 7,
                                 8, 9, 10, 11, 12, 13, 14, 15);
}

static __device__ __forceinline__ v8f wmma_bf16(v16bf a, v16bf b, v8f c) {
  // D = A(16x32 bf16) x B(32x16 bf16) + C(16x16 f32)
  return __builtin_amdgcn_wmma_f32_16x16x32_bf16(false, a, false, b,
                                                 (short)0, c, false, false);
}

// 16-byte global -> LDS copy; async (ASYNCcnt) when the toolchain has it.
static __device__ __forceinline__ void lds_copy16(const void* g, void* l) {
#ifdef USE_ASYNC_LDS
  __builtin_amdgcn_global_load_async_to_lds_b128((as1_v4i)g, (as3_v4i)l, 0, 0);
#else
  *reinterpret_cast<bf16x8*>(l) = *reinterpret_cast<const bf16x8*>(g);
#endif
}

static __device__ __forceinline__ void lds_copy_wait() {
#ifdef USE_ASYNC_LDS
#if __has_builtin(__builtin_amdgcn_s_wait_asynccnt)
  __builtin_amdgcn_s_wait_asynccnt(0);
#else
  asm volatile("s_wait_asynccnt 0x0" ::: "memory");
#endif
#endif
}

// ---------------------------------------------------------------------------
// fp32 -> bf16 conversion
// ---------------------------------------------------------------------------
__global__ __launch_bounds__(256) void f32_to_bf16_kernel(
    const float* __restrict__ src, bf16* __restrict__ dst, long n) {
  long i = (long)blockIdx.x * 256 + threadIdx.x;
  const long stride = (long)gridDim.x * 256;
  for (; i < n; i += stride) dst[i] = (bf16)src[i];
}

// ---------------------------------------------------------------------------
// RMSNorm: one block per row; fp32 in, bf16 out
// ---------------------------------------------------------------------------
__global__ __launch_bounds__(256) void rmsnorm_kernel(
    const float* __restrict__ x, const float* __restrict__ w,
    bf16* __restrict__ out) {
  __shared__ float red[256];
  const int s = blockIdx.x, tid = threadIdx.x;
  const float* xr = x + (size_t)s * ACH;
  float ss = 0.f;
  for (int i = tid * 4; i < ACH; i += 1024) {
    f32x4 v = *reinterpret_cast<const f32x4*>(xr + i);
    ss += v[0] * v[0] + v[1] * v[1] + v[2] * v[2] + v[3] * v[3];
  }
  red[tid] = ss;
  __syncthreads();
  for (int off = 128; off > 0; off >>= 1) {
    if (tid < off) red[tid] += red[tid + off];
    __syncthreads();
  }
  const float scale = rsqrtf(red[0] / (float)ACH + 1e-6f);
  bf16* orow = out + (size_t)s * ACH;
  for (int i = tid * 4; i < ACH; i += 1024) {
    f32x4 v = *reinterpret_cast<const f32x4*>(xr + i);
    f32x4 g = *reinterpret_cast<const f32x4*>(w + i);
    bf16x4 o;
    o[0] = (bf16)(v[0] * scale * g[0]);
    o[1] = (bf16)(v[1] * scale * g[1]);
    o[2] = (bf16)(v[2] * scale * g[2]);
    o[3] = (bf16)(v[3] * scale * g[3]);
    *reinterpret_cast<bf16x4*>(orow + i) = o;
  }
}

// ---------------------------------------------------------------------------
// Generic WMMA GEMM: OUT[M,N] = A[M,K](bf16) x W[N,K](bf16)^T
// Block tile 256x64 (8 waves x 32 rows), LDS-staged W tile, K step 32.
// Each B fragment feeds two A fragments -> 8 WMMA per staged tile per wave.
// OUTMODE: 0 = fp32 store, 1 = bf16 store, 2 = fp32 + residual add
// ---------------------------------------------------------------------------
#define G_WSTRIDE 40  // bf16 elems per LDS row: 32 data + 8 pad (conflict-free)

template <int OUTMODE>
__global__ __launch_bounds__(256) void gemm_bf16_kernel(
    const bf16* __restrict__ A, const bf16* __restrict__ W,
    float* __restrict__ outF, bf16* __restrict__ outB,
    const float* __restrict__ resid, int M, int N, int K) {
  __shared__ bf16 Wt[64 * G_WSTRIDE];
  const int tid = threadIdx.x;
  const int lane = tid & 31, wave = tid >> 5;
  const int hx = lane >> 4, lr = lane & 15;
  const int n0 = blockIdx.x * 64;
  const int m0 = blockIdx.y * 256 + wave * 32;

  const int srow = tid >> 2;      // 0..63 : W row (N) being staged
  const int skk = (tid & 3) * 8;  // k segment within 32
  const bf16* Wp = W + (size_t)(n0 + srow) * K + skk;
  const bf16* Ap0 = A + (size_t)(m0 + lr) * K;
  const bf16* Ap1 = A + (size_t)(m0 + 16 + lr) * K;

  v8f acc[2][4] = {};
  for (int k0 = 0; k0 < K; k0 += 32) {
    __syncthreads();  // previous tile fully consumed
    lds_copy16(Wp + k0, &Wt[srow * G_WSTRIDE + skk]);
    lds_copy_wait();
    __syncthreads();
    if (k0 + 32 < K) __builtin_prefetch(Wp + k0 + 32, 0, 1);
    // A fragments: lane=row, elems K = {8*hx+0..7, 16+8*hx+0..7}
    v16bf af0 = mk_frag(*(const bf16x8*)(Ap0 + k0 + hx * 8),
                        *(const bf16x8*)(Ap0 + k0 + 16 + hx * 8));
    v16bf af1 = mk_frag(*(const bf16x8*)(Ap1 + k0 + hx * 8),
                        *(const bf16x8*)(Ap1 + k0 + 16 + hx * 8));
#pragma unroll
    for (int f = 0; f < 4; ++f) {
      // B fragment: lane=col, elems K = {8*hx+0..7, 16+8*hx+0..7}
      const bf16* wl = &Wt[(f * 16 + lr) * G_WSTRIDE + hx * 8];
      v16bf bfr = mk_frag(*(const bf16x8*)(wl), *(const bf16x8*)(wl + 16));
      acc[0][f] = wmma_bf16(af0, bfr, acc[0][f]);
      acc[1][f] = wmma_bf16(af1, bfr, acc[1][f]);
    }
  }
#pragma unroll
  for (int sub = 0; sub < 2; ++sub) {
#pragma unroll
    for (int f = 0; f < 4; ++f) {
#pragma unroll
      for (int g = 0; g < 8; ++g) {
        const int row = m0 + sub * 16 + g + hx * 8;
        const size_t idx = (size_t)row * N + n0 + f * 16 + lr;
        float v = acc[sub][f][g];
        if constexpr (OUTMODE == 2) v += resid[idx];
        if constexpr (OUTMODE == 1)
          outB[idx] = (bf16)v;
        else
          outF[idx] = v;
      }
    }
  }
}

// ---------------------------------------------------------------------------
// RoPE (fp32 -> bf16), optionally emitting per-(head,key) sum of squares k2
// one wave per (s, h)
// ---------------------------------------------------------------------------
__global__ __launch_bounds__(256) void rope_kernel(
    const float* __restrict__ x, const float* __restrict__ cosT,
    const float* __restrict__ sinT, bf16* __restrict__ out,
    float* __restrict__ k2) {
  const int wid = blockIdx.x * 8 + (threadIdx.x >> 5);
  const int lane = threadIdx.x & 31;
  const int s = wid >> 4, h = wid & 15;
  const size_t base = ((size_t)s * AH + h) * AD;
  const float* cr = cosT + (size_t)s * AD;
  const float* sr = sinT + (size_t)s * AD;
  float acc = 0.f;
#pragma unroll
  for (int p = lane; p < AD / 2; p += 32) {
    float x1 = x[base + p], x2 = x[base + p + AD / 2];
    float r1 = x1 * cr[p] - x2 * sr[p];
    float r2 = x2 * cr[p + AD / 2] + x1 * sr[p + AD / 2];
    out[base + p] = (bf16)r1;
    out[base + p + AD / 2] = (bf16)r2;
    acc += r1 * r1 + r2 * r2;
  }
  if (k2) {
#pragma unroll
    for (int off = 16; off > 0; off >>= 1) acc += __shfl_xor(acc, off, 32);
    if (lane == 0) k2[(size_t)h * ASEQ + s] = acc;
  }
}

// ---------------------------------------------------------------------------
// Flash attention with Gaussian-kernel (W2) scores, all-WMMA.
// scores^T = K x Q^T  (softmax row lives in a lane pair)
// O^T      = V^T x P^T (V transposed through LDS)
// grid: (H, S/64), block 128 (4 waves x 16 q rows)
// ---------------------------------------------------------------------------
#define FA_KS 136  // K tile LDS row stride (128 + 8 pad)
#define FA_VS 40   // V^T tile LDS row stride (32 + 8 pad)

__global__ __launch_bounds__(128) void attention_kernel(
    const bf16* __restrict__ Q, const bf16* __restrict__ Km,
    const bf16* __restrict__ V, const float* __restrict__ k2,
    bf16* __restrict__ ctx) {
  __shared__ bf16 Ks[32 * FA_KS];
  __shared__ bf16 Vt[AD * FA_VS];
  const int tid = threadIdx.x;
  const int lane = tid & 31, wave = tid >> 5;
  const int hx = lane >> 4, lr = lane & 15;
  const int h = blockIdx.x;
  const int qb = blockIdx.y * 64;
  const int qrow = qb + wave * 16 + lr;

  // Q fragments as B operand: lane=q col, elems K(d) = dc*32 + 16*hx + 0..15
  const bf16* qp = Q + ((size_t)qrow * AH + h) * AD;
  v16bf qf[4];
#pragma unroll
  for (int dc = 0; dc < 4; ++dc) {
    bf16x8 lo = *reinterpret_cast<const bf16x8*>(qp + dc * 32 + hx * 16);
    bf16x8 hi = *reinterpret_cast<const bf16x8*>(qp + dc * 32 + hx * 16 + 8);
    qf[dc] = mk_frag(lo, hi);
  }

  v8f o[8] = {};
  float m = -3.0e38f, l = 0.f;

  const int kt = tid & 31;            // staged key
  const int dbase = (tid >> 5) * 32;  // staged d segment
  const int nchunk = (qb + 64) >> 5;  // causal: keys <= qb+63

  for (int ch = 0; ch < nchunk; ++ch) {
    const int kb = ch * 32;
    __syncthreads();
    const size_t srcb = ((size_t)(kb + kt) * AH + h) * AD + dbase;
#pragma unroll
    for (int seg = 0; seg < 4; ++seg) {
      lds_copy16(Km + srcb + seg * 8, &Ks[kt * FA_KS + dbase + seg * 8]);
      bf16x8 vv = *reinterpret_cast<const bf16x8*>(V + srcb + seg * 8);
#pragma unroll
      for (int e = 0; e < 8; ++e)
        Vt[(dbase + seg * 8 + e) * FA_VS + kt] = vv[e];
    }
    lds_copy_wait();
    __syncthreads();

    // ---- scores^T tiles (keys x q), two 16-key tiles per 32-key chunk ----
    float sc0[8], sc1[8];
    float mloc = -3.0e38f;
#pragma unroll
    for (int t = 0; t < 2; ++t) {
      v8f c = {};
#pragma unroll
      for (int dc = 0; dc < 4; ++dc) {
        // A fragment: lane=key row, elems K(d) = dc*32 + 16*hx + 0..15
        const bf16* kl = &Ks[(t * 16 + lr) * FA_KS + dc * 32 + hx * 16];
        bf16x8 lo = *reinterpret_cast<const bf16x8*>(kl);
        bf16x8 hi = *reinterpret_cast<const bf16x8*>(kl + 8);
        c = wmma_bf16(mk_frag(lo, hi), qf[dc], c);
      }
      // lane's keys: kb + t*16 + 8*hx + g   (C layout: M = g + 8*hx)
      const float* k2p = k2 + (size_t)h * ASEQ + kb + t * 16 + hx * 8;
      f32x4 k2lo = *reinterpret_cast<const f32x4*>(k2p);
      f32x4 k2hi = *reinterpret_cast<const f32x4*>(k2p + 4);
      float* sc = t ? sc1 : sc0;
#pragma unroll
      for (int g = 0; g < 8; ++g) {
        const int key = kb + t * 16 + hx * 8 + g;
        const float k2v = (g < 4) ? k2lo[g] : k2hi[g - 4];
        // -(q2 - 2qk + k2)/(2*sqrt(D)); q2 is softmax-row-constant -> dropped
        float v = (2.f * c[g] - k2v) * INV_2SQRTD;
        if (key > qrow) v = -1.0e9f;  // causal additive mask
        sc[g] = v;
        mloc = fmaxf(mloc, v);
      }
    }

    // ---- online softmax (row = lane pair L, L^16) ----
    const float mo = fmaxf(mloc, __shfl_xor(mloc, 16, 32));
    const float mnew = fmaxf(m, mo);
    const float alpha = __expf(m - mnew);
    m = mnew;
#pragma unroll
    for (int i = 0; i < 8; ++i) o[i] *= alpha;
    float lsum = 0.f;
    float p0[8], p1[8];
#pragma unroll
    for (int g = 0; g < 8; ++g) {
      p0[g] = __expf(sc0[g] - mnew);
      p1[g] = __expf(sc1[g] - mnew);
      lsum += p0[g] + p1[g];
    }
    l = l * alpha + lsum;

    // ---- pack P^T into B-fragment layout: lane=q, elems K(key)=16*hx+0..15
    // hx=0 needs keys 0..15 (tile0), hx=1 needs keys 16..31 (tile1);
    // each lane owns half of each tile -> exchange with partner lane (^16).
    v16bf pf = {};
#pragma unroll
    for (int g = 0; g < 8; ++g) {
      const float p0p = __shfl_xor(p0[g], 16, 32);
      const float p1p = __shfl_xor(p1[g], 16, 32);
      pf[g] = (bf16)(hx ? p1p : p0[g]);
      pf[g + 8] = (bf16)(hx ? p1[g] : p0p);
    }

    // ---- O^T += V^T x P^T : 8 d-slices of 16 ----
#pragma unroll
    for (int ds = 0; ds < 8; ++ds) {
      // A fragment: lane = d row, elems K(key) = 16*hx + 0..15
      const bf16* vl = &Vt[(ds * 16 + lr) * FA_VS + hx * 16];
      bf16x8 lo = *reinterpret_cast<const bf16x8*>(vl);
      bf16x8 hi = *reinterpret_cast<const bf16x8*>(vl + 8);
      o[ds] = wmma_bf16(mk_frag(lo, hi), pf, o[ds]);
    }
  }

  l += __shfl_xor(l, 16, 32);
  const float rl = (l > 0.f) ? 1.f / l : 0.f;
  // O^T C layout: lane q = lr, d = ds*16 + g + 8*hx (g contiguous)
  bf16* op = ctx + ((size_t)qrow * AH + h) * AD;
#pragma unroll
  for (int ds = 0; ds < 8; ++ds) {
    bf16x8 ov;
#pragma unroll
    for (int g = 0; g < 8; ++g) ov[g] = (bf16)(o[ds][g] * rl);
    *reinterpret_cast<bf16x8*>(op + ds * 16 + hx * 8) = ov;
  }
}

// ---------------------------------------------------------------------------
// SwiGLU elementwise: t = silu(g) * u  (fp32 -> bf16)
// ---------------------------------------------------------------------------
__global__ __launch_bounds__(256) void swiglu_kernel(
    const float* __restrict__ g, const float* __restrict__ u,
    bf16* __restrict__ t, long n) {
  long i = (long)blockIdx.x * 256 + threadIdx.x;
  const long stride = (long)gridDim.x * 256;
  for (; i < n; i += stride) {
    const float gv = g[i];
    const float s = gv / (1.f + __expf(-gv));
    t[i] = (bf16)(s * u[i]);
  }
}

// ---------------------------------------------------------------------------
extern "C" void kernel_launch(void* const* d_in, const int* in_sizes, int n_in,
                              void* d_out, int out_size, void* d_ws,
                              size_t ws_size, hipStream_t stream) {
  const float* hs   = (const float*)d_in[0];
  const float* cosT = (const float*)d_in[1];
  const float* sinT = (const float*)d_in[2];
  // d_in[3]: attention_mask — causal, applied analytically in-kernel
  const float* ln1 = (const float*)d_in[4];
  const float* wq  = (const float*)d_in[5];
  const float* wk  = (const float*)d_in[6];
  const float* wv  = (const float*)d_in[7];
  const float* wo  = (const float*)d_in[8];
  const float* ln2 = (const float*)d_in[9];
  const float* wg  = (const float*)d_in[10];
  const float* wu  = (const float*)d_in[11];
  const float* wd  = (const float*)d_in[12];

  char* ws = (char*)d_ws;
  size_t off = 0;
  auto alloc = [&](size_t bytes) -> void* {
    void* p = ws + off;
    off += (bytes + 255) & ~(size_t)255;
    return p;
  };

  const size_t CC = (size_t)ACH * ACH;
  const size_t CF = (size_t)ACH * AFF;
  const size_t SC = (size_t)ASEQ * ACH;
  const size_t SF = (size_t)ASEQ * AFF;

  bf16* wq_b = (bf16*)alloc(CC * 2);
  bf16* wk_b = (bf16*)alloc(CC * 2);
  bf16* wv_b = (bf16*)alloc(CC * 2);
  bf16* wo_b = (bf16*)alloc(CC * 2);
  bf16* wg_b = (bf16*)alloc(CF * 2);
  bf16* wu_b = (bf16*)alloc(CF * 2);
  bf16* wd_b = (bf16*)alloc(CF * 2);

  const size_t attn_region = off;  // reused by FFN gate buffer afterwards
  bf16*  xnorm = (bf16*)alloc(SC * 2);
  float* qf    = (float*)alloc(SC * 4);
  float* kf    = (float*)alloc(SC * 4);
  bf16*  q_bf  = (bf16*)alloc(SC * 2);
  bf16*  k_bf  = (bf16*)alloc(SC * 2);
  bf16*  v_bf  = (bf16*)alloc(SC * 2);

  float* k2b  = (float*)alloc((size_t)AH * ASEQ * 4);
  bf16*  ctx  = (bf16*)alloc(SC * 2);
  float* hbuf = (float*)alloc(SC * 4);
  bf16*  ybuf = (bf16*)alloc(SC * 2);
  float* ubuf = (float*)alloc(SF * 4);
  bf16*  tbuf = (bf16*)alloc(SF * 2);
  float* gbuf = (float*)(ws + attn_region);  // alias: attn buffers dead by then

  // --- weight conversion fp32 -> bf16 ---
  f32_to_bf16_kernel<<<2048, 256, 0, stream>>>(wq, wq_b, (long)CC);
  f32_to_bf16_kernel<<<2048, 256, 0, stream>>>(wk, wk_b, (long)CC);
  f32_to_bf16_kernel<<<2048, 256, 0, stream>>>(wv, wv_b, (long)CC);
  f32_to_bf16_kernel<<<2048, 256, 0, stream>>>(wo, wo_b, (long)CC);
  f32_to_bf16_kernel<<<2048, 256, 0, stream>>>(wg, wg_b, (long)CF);
  f32_to_bf16_kernel<<<2048, 256, 0, stream>>>(wu, wu_b, (long)CF);
  f32_to_bf16_kernel<<<2048, 256, 0, stream>>>(wd, wd_b, (long)CF);

  // --- attention path ---
  rmsnorm_kernel<<<ASEQ, 256, 0, stream>>>(hs, ln1, xnorm);

  dim3 gqkv(ACH / 64, ASEQ / 256);
  gemm_bf16_kernel<0><<<gqkv, 256, 0, stream>>>(xnorm, wq_b, qf, nullptr,
                                                nullptr, ASEQ, ACH, ACH);
  gemm_bf16_kernel<0><<<gqkv, 256, 0, stream>>>(xnorm, wk_b, kf, nullptr,
                                                nullptr, ASEQ, ACH, ACH);
  gemm_bf16_kernel<1><<<gqkv, 256, 0, stream>>>(xnorm, wv_b, nullptr, v_bf,
                                                nullptr, ASEQ, ACH, ACH);

  rope_kernel<<<(ASEQ * AH) / 8, 256, 0, stream>>>(qf, cosT, sinT, q_bf,
                                                   nullptr);
  rope_kernel<<<(ASEQ * AH) / 8, 256, 0, stream>>>(kf, cosT, sinT, k_bf, k2b);

  attention_kernel<<<dim3(AH, ASEQ / 64), 128, 0, stream>>>(q_bf, k_bf, v_bf,
                                                            k2b, ctx);

  gemm_bf16_kernel<2><<<gqkv, 256, 0, stream>>>(ctx, wo_b, hbuf, nullptr, hs,
                                                ASEQ, ACH, ACH);

  // --- FFN path ---
  rmsnorm_kernel<<<ASEQ, 256, 0, stream>>>(hbuf, ln2, ybuf);

  dim3 gff(AFF / 64, ASEQ / 256);
  gemm_bf16_kernel<0><<<gff, 256, 0, stream>>>(ybuf, wg_b, gbuf, nullptr,
                                               nullptr, ASEQ, AFF, ACH);
  gemm_bf16_kernel<0><<<gff, 256, 0, stream>>>(ybuf, wu_b, ubuf, nullptr,
                                               nullptr, ASEQ, AFF, ACH);
  swiglu_kernel<<<2048, 256, 0, stream>>>(gbuf, ubuf, tbuf, (long)SF);

  gemm_bf16_kernel<2><<<gqkv, 256, 0, stream>>>(tbuf, wd_b, (float*)d_out,
                                                nullptr, hbuf, ASEQ, ACH, AFF);
}
